// motivate_model_50302656971537
// MI455X (gfx1250) — matrix operand order repaired and checked by reading, not compiled, not run on hardware
//
#include <hip/hip_runtime.h>

// ---------------------------------------------------------------------------
// MI455X (gfx1250, wave32) implementation.
// All GEMMs run on v_wmma_f32_16x16x32_bf16 (bf16 in, f32 accumulate).
// Weights are packed once into the WMMA B-fragment layout so the recurrent
// hot loop issues only contiguous 128-bit loads + WMMA. The recurrent cell
// kernel is explicitly software-pipelined (double-buffered A and B fragments);
// __launch_bounds__(...,1) lifts the VGPR budget so the double buffer is not
// coalesced away (the B=256 recurrence has too few waves for occupancy-based
// latency hiding, so ILP is everything).
// ---------------------------------------------------------------------------

typedef __attribute__((ext_vector_type(16))) __bf16 v16bf;
typedef __attribute__((ext_vector_type(8)))  float  v8f;

#define T_  96
#define B_  256
#define E_  128
#define H_  256
#define G7  1792    // 7*H
#define P0_ 128
#define P1_ 64

// ---------------- scalar helpers ----------------
__device__ __forceinline__ unsigned short f2bf(float f) {
    unsigned u = __builtin_bit_cast(unsigned, f);
    unsigned r = (u + 0x7FFFu + ((u >> 16) & 1u)) >> 16;   // RNE
    return (unsigned short)r;
}
__device__ __forceinline__ float bf2f(unsigned short s) {
    return __builtin_bit_cast(float, (unsigned)s << 16);
}
__device__ __forceinline__ float sigf(float x)  { return 1.0f / (1.0f + __expf(-x)); }
__device__ __forceinline__ float tanhfast(float x) { return 2.0f * sigf(2.0f * x) - 1.0f; }
__device__ __forceinline__ float leaky(float x) { return x > 0.0f ? x : 0.3f * x; }

// ---------------- WMMA fragment loaders ----------------
// A fragment (16x32, bf16) from row-major [M x ldk] bf16 (ISA 7.12.2):
// lanes 0-15: row m, K {0..7,16..23}; lanes 16-31: row m, K {8..15,24..31}.
__device__ __forceinline__ v16bf load_a_frag(const unsigned short* base, int ldk,
                                             int m0, int k0) {
    int lane = threadIdx.x & 31;
    const unsigned short* r = base + (size_t)(m0 + (lane & 15)) * ldk + k0
                            + ((lane & 16) ? 8 : 0);
    union { v16bf v; uint4 q[2]; } u;
    u.q[0] = *(const uint4*)(r);
    u.q[1] = *(const uint4*)(r + 16);
    return u.v;
}

// B fragment (32x16, bf16) from the packed stream: fragment f occupies 512
// bf16; lane's 16 values are contiguous (32 B) => two b128 loads.
__device__ __forceinline__ v16bf load_b_frag(const unsigned short* pk, int frag) {
    int lane = threadIdx.x & 31;
    const unsigned short* p = pk + (size_t)frag * 512 + lane * 16;
    union { v16bf v; uint4 q[2]; } u;
    u.q[0] = *(const uint4*)(p);
    u.q[1] = *(const uint4*)(p + 8);
    return u.v;
}

__device__ __forceinline__ v8f wmma_bf16(v16bf a, v16bf b, v8f c) {
    return __builtin_amdgcn_wmma_f32_16x16x32_bf16(
        false, a, false, b, (short)0, c, false, false);
}

// ---------------- prep kernels ----------------
// Pack row-major fp32 W[K x N] into B-fragment layout, kt innermost:
// frag = nt*(K/32)+kt; within a frag, lane l<16 holds col nt*16+l, K kt*32+{0..7,16..23};
// lane l>=16 holds col nt*16+(l-16), K kt*32+{8..15,24..31}.
__global__ void pack_b_kernel(const float* __restrict__ W,
                              unsigned short* __restrict__ out, int K, int N) {
    int idx = blockIdx.x * blockDim.x + threadIdx.x;
    if (idx >= K * N) return;
    int f    = idx >> 9;
    int r    = idx & 511;
    int lane = r >> 4;
    int j    = r & 15;
    int KT   = K >> 5;
    int nt   = f / KT;
    int kt   = f - nt * KT;
    int krow = kt * 32 + ((lane & 16) ? 8 : 0) + j + ((j & 8) ? 8 : 0);
    int col  = nt * 16 + (lane & 15);
    out[idx] = f2bf(W[(size_t)krow * N + col]);
}

__global__ void cvt_x_kernel(const float* __restrict__ x,
                             unsigned short* __restrict__ xb, int n) {
    int idx = blockIdx.x * blockDim.x + threadIdx.x;
    if (idx < n) xb[idx] = f2bf(x[idx]);
}

__global__ void zero_state_kernel(unsigned short* hC0, unsigned short* hV0,
                                  float* cC, float* chC, float* cV, float* chV) {
    int idx = blockIdx.x * blockDim.x + threadIdx.x;
    if (idx >= B_ * H_) return;
    hC0[idx] = 0; hV0[idx] = 0;
    cC[idx] = 0.0f; chC[idx] = 0.0f; cV[idx] = 0.0f; chV[idx] = 0.0f;
}

// ---------------- recurrent cell kernel ----------------
// grid (16 h-tiles, 4 batch-tile groups, 2 cells), block 128 (4 waves).
// Each wave: 16(batch) x 16(h) tile, 7 gate accumulators. A unified 12-step
// K pipeline (4 steps x@Wx with K=128, then 8 steps h@Wh with K=256), with
// next-step A/B fragments prefetched into a second register buffer so the
// 16 b128 loads of step s+1 overlap the 7 WMMAs of step s.
__global__ void __launch_bounds__(128, 1)
cell_kernel(const unsigned short* __restrict__ Xbf,
            const unsigned short* __restrict__ WxPkC,
            const unsigned short* __restrict__ WhPkC,
            const float* __restrict__ bxC,
            const unsigned short* __restrict__ WxPkV,
            const unsigned short* __restrict__ WhPkV,
            const float* __restrict__ bxV,
            const unsigned short* __restrict__ hInC,
            unsigned short* __restrict__ hOutC,
            float* __restrict__ cC, float* __restrict__ chC,
            const unsigned short* __restrict__ hInV,
            unsigned short* __restrict__ hOutV,
            float* __restrict__ cV, float* __restrict__ chV,
            int t) {
    const int wave = threadIdx.x >> 5;
    const int bt   = blockIdx.y * 4 + wave;
    const int ht   = blockIdx.x;
    const int cell = blockIdx.z;

    const unsigned short* WxPk = cell ? WxPkV : WxPkC;
    const unsigned short* WhPk = cell ? WhPkV : WhPkC;
    const float*          bx   = cell ? bxV   : bxC;
    const unsigned short* hIn  = cell ? hInV  : hInC;
    unsigned short*       hOut = cell ? hOutV : hOutC;
    float*                cS   = cell ? cV    : cC;
    float*                chS  = cell ? chV   : chC;

    const unsigned short* xrow = Xbf + (size_t)t * B_ * E_;

    v8f acc[7];
#pragma unroll
    for (int g = 0; g < 7; ++g) acc[g] = (v8f){0,0,0,0,0,0,0,0};

    // fragment fetch for unified k-step s in [0,12)
    auto load_step = [&](int s, v16bf& a, v16bf (&b)[7]) {
        if (s < E_ / 32) {                       // x @ Wx phase
            a = load_a_frag(xrow, E_, bt * 16, s * 32);
#pragma unroll
            for (int g = 0; g < 7; ++g)
                b[g] = load_b_frag(WxPk, (g * 16 + ht) * (E_ / 32) + s);
        } else {                                 // h @ Wh phase
            const int kt = s - E_ / 32;
            a = load_a_frag(hIn, H_, bt * 16, kt * 32);
#pragma unroll
            for (int g = 0; g < 7; ++g)
                b[g] = load_b_frag(WhPk, (g * 16 + ht) * (H_ / 32) + kt);
        }
    };

    constexpr int NS = E_ / 32 + H_ / 32;        // 12 k-steps
    v16bf a_cur, a_nxt;
    v16bf b_cur[7], b_nxt[7];
    load_step(0, a_cur, b_cur);
#pragma unroll
    for (int s = 0; s < NS; ++s) {
        if (s + 1 < NS) load_step(s + 1, a_nxt, b_nxt);
#pragma unroll
        for (int g = 0; g < 7; ++g)
            acc[g] = wmma_bf16(a_cur, b_cur[g], acc[g]);
        a_cur = a_nxt;
#pragma unroll
        for (int g = 0; g < 7; ++g) b_cur[g] = b_nxt[g];
    }

    // gate epilogue. C/D layout: VGPR j, lanes 0-15 -> M=j; lanes 16-31 -> M=j+8.
    const int lane = threadIdx.x & 31;
    const int n    = lane & 15;
    const int mh   = (lane & 16) ? 8 : 0;
    const int hidx = ht * 16 + n;

    float bias[7];
#pragma unroll
    for (int g = 0; g < 7; ++g) bias[g] = bx[g * H_ + hidx];

#pragma unroll
    for (int j = 0; j < 8; ++j) {
        const int bidx = bt * 16 + j + mh;
        const size_t s = (size_t)bidx * H_ + hidx;
        float gi  = sigf(acc[0][j] + bias[0]);
        float gih = sigf(acc[1][j] + bias[1]);
        float gf  = sigf(acc[2][j] + bias[2]);
        float gfh = sigf(acc[3][j] + bias[3]);
        float go  = sigf(acc[4][j] + bias[4]);
        float gz  = tanhfast(acc[5][j] + bias[5]);
        float gt  = 5.0f * sigf((acc[6][j] + bias[6]) * 0.2f);
        float c   = cS[s];
        float ch  = chS[s];
        float ct  = ch + (c - ch) * __expf(-gt);
        float hn  = go * tanhfast(ct);
        cS[s]  = gf * ct + gi * gz;
        chS[s] = gfh * ch + gih * gz;
        hOut[s] = f2bf(hn);
    }
}

// ---------------- prediction-head kernel ----------------
// grid (2 cells), block 256 (8 waves). y0 (256x128 bf16) and y1 (256x64 bf16)
// live in LDS (96 KB of the 320 KB WGP pool).
__global__ void __launch_bounds__(256, 1)
head_kernel(const unsigned short* __restrict__ hC,
            const unsigned short* __restrict__ hV,
            const unsigned short* __restrict__ W0CPk,
            const float* __restrict__ b0C,
            const unsigned short* __restrict__ W1CPk,
            const float* __restrict__ b1C,
            const float* __restrict__ fcCW,
            const float* __restrict__ fcCb,
            const unsigned short* __restrict__ W0VPk,
            const float* __restrict__ b0V,
            const unsigned short* __restrict__ W1VPk,
            const float* __restrict__ b1V,
            const float* __restrict__ fcVW,
            const float* __restrict__ fcVb,
            float* __restrict__ pC, float* __restrict__ pV,
            int t) {
    extern __shared__ unsigned short smem[];
    unsigned short* y0 = smem;                 // 256*128 bf16 = 64 KB
    unsigned short* y1 = smem + B_ * P0_;      // 256*64  bf16 = 32 KB

    const int cell = blockIdx.x;
    const unsigned short* h    = cell ? hV    : hC;
    const unsigned short* W0Pk = cell ? W0VPk : W0CPk;
    const float*          b0   = cell ? b0V   : b0C;
    const unsigned short* W1Pk = cell ? W1VPk : W1CPk;
    const float*          b1   = cell ? b1V   : b1C;
    const float*          fcW  = cell ? fcVW  : fcCW;
    const float*          fcb  = cell ? fcVb  : fcCb;
    float*                pOut = cell ? pV    : pC;

    const int wave = threadIdx.x >> 5;
    const int lane = threadIdx.x & 31;
    const int n    = lane & 15;
    const int mh   = (lane & 16) ? 8 : 0;

    // layer 1: y0 = leaky(h @ W0 + b0)   [256x256]@[256x128]
#pragma unroll
    for (int mi = 0; mi < 2; ++mi) {
        const int mt = wave * 2 + mi;
        v8f acc[P0_ / 16];
#pragma unroll
        for (int nt = 0; nt < P0_ / 16; ++nt) acc[nt] = (v8f){0,0,0,0,0,0,0,0};
#pragma unroll
        for (int kt = 0; kt < H_ / 32; ++kt) {
            v16bf a = load_a_frag(h, H_, mt * 16, kt * 32);
            v16bf bfr[P0_ / 16];
#pragma unroll
            for (int nt = 0; nt < P0_ / 16; ++nt)
                bfr[nt] = load_b_frag(W0Pk, nt * (H_ / 32) + kt);
#pragma unroll
            for (int nt = 0; nt < P0_ / 16; ++nt)
                acc[nt] = wmma_bf16(a, bfr[nt], acc[nt]);
        }
#pragma unroll
        for (int nt = 0; nt < P0_ / 16; ++nt) {
            float bb = b0[nt * 16 + n];
#pragma unroll
            for (int j = 0; j < 8; ++j) {
                int row = mt * 16 + j + mh;
                y0[row * P0_ + nt * 16 + n] = f2bf(leaky(acc[nt][j] + bb));
            }
        }
    }
    __syncthreads();

    // layer 2: y1 = leaky(y0 @ W1 + b1)  [256x128]@[128x64]
#pragma unroll
    for (int mi = 0; mi < 2; ++mi) {
        const int mt = wave * 2 + mi;
        v8f acc[P1_ / 16];
#pragma unroll
        for (int nt = 0; nt < P1_ / 16; ++nt) acc[nt] = (v8f){0,0,0,0,0,0,0,0};
#pragma unroll
        for (int kt = 0; kt < P0_ / 32; ++kt) {
            v16bf a = load_a_frag(y0, P0_, mt * 16, kt * 32);
            v16bf bfr[P1_ / 16];
#pragma unroll
            for (int nt = 0; nt < P1_ / 16; ++nt)
                bfr[nt] = load_b_frag(W1Pk, nt * (P0_ / 32) + kt);
#pragma unroll
            for (int nt = 0; nt < P1_ / 16; ++nt)
                acc[nt] = wmma_bf16(a, bfr[nt], acc[nt]);
        }
#pragma unroll
        for (int nt = 0; nt < P1_ / 16; ++nt) {
            float bb = b1[nt * 16 + n];
#pragma unroll
            for (int j = 0; j < 8; ++j) {
                int row = mt * 16 + j + mh;
                y1[row * P1_ + nt * 16 + n] = f2bf(leaky(acc[nt][j] + bb));
            }
        }
    }
    __syncthreads();

    // layer 3: p = sigmoid(y1 @ fcW + fcb)   (dot-64 per batch row)
    {
        const int b = threadIdx.x;   // 256 threads == 256 rows
        float s = fcb[0];
#pragma unroll
        for (int k = 0; k < P1_; ++k) s += bf2f(y1[b * P1_ + k]) * fcW[k];
        pOut[(size_t)t * B_ + b] = sigf(s);
    }
}

__global__ void finalize_kernel(const float* __restrict__ pC,
                                const float* __restrict__ pV,
                                float* __restrict__ out) {
    int idx = blockIdx.x * blockDim.x + threadIdx.x;
    if (idx >= T_ * B_) return;
    out[idx]           = pC[idx];
    out[T_ * B_ + idx] = pC[idx] * pV[idx];
}

// ---------------------------------------------------------------------------
extern "C" void kernel_launch(void* const* d_in, const int* in_sizes, int n_in,
                              void* d_out, int out_size, void* d_ws, size_t ws_size,
                              hipStream_t stream) {
    const float* x     = (const float*)d_in[0];
    const float* Wx_c  = (const float*)d_in[1];
    const float* bx_c  = (const float*)d_in[2];
    const float* Wh_c  = (const float*)d_in[3];
    const float* Wx_v  = (const float*)d_in[4];
    const float* bx_v  = (const float*)d_in[5];
    const float* Wh_v  = (const float*)d_in[6];
    const float* pcW0  = (const float*)d_in[7];
    const float* pcb0  = (const float*)d_in[8];
    const float* pcW1  = (const float*)d_in[9];
    const float* pcb1  = (const float*)d_in[10];
    const float* pvW0  = (const float*)d_in[11];
    const float* pvb0  = (const float*)d_in[12];
    const float* pvW1  = (const float*)d_in[13];
    const float* pvb1  = (const float*)d_in[14];
    const float* fccW  = (const float*)d_in[15];
    const float* fccb  = (const float*)d_in[16];
    const float* fcvW  = (const float*)d_in[17];
    const float* fcvb  = (const float*)d_in[18];
    float* out = (float*)d_out;

    // bump allocator over d_ws (total ~11 MB)
    char* ws = (char*)d_ws;
    size_t off = 0;
    auto alloc = [&](size_t bytes) -> char* {
        off = (off + 255) & ~(size_t)255;
        char* p = ws + off;
        off += bytes;
        return p;
    };
    unsigned short* Xbf   = (unsigned short*)alloc((size_t)T_ * B_ * E_ * 2);
    unsigned short* WxPkC = (unsigned short*)alloc((size_t)E_ * G7 * 2);
    unsigned short* WxPkV = (unsigned short*)alloc((size_t)E_ * G7 * 2);
    unsigned short* WhPkC = (unsigned short*)alloc((size_t)H_ * G7 * 2);
    unsigned short* WhPkV = (unsigned short*)alloc((size_t)H_ * G7 * 2);
    unsigned short* W0PkC = (unsigned short*)alloc((size_t)H_ * P0_ * 2);
    unsigned short* W0PkV = (unsigned short*)alloc((size_t)H_ * P0_ * 2);
    unsigned short* W1PkC = (unsigned short*)alloc((size_t)P0_ * P1_ * 2);
    unsigned short* W1PkV = (unsigned short*)alloc((size_t)P0_ * P1_ * 2);
    unsigned short* hC[2] = {(unsigned short*)alloc((size_t)B_ * H_ * 2),
                             (unsigned short*)alloc((size_t)B_ * H_ * 2)};
    unsigned short* hV[2] = {(unsigned short*)alloc((size_t)B_ * H_ * 2),
                             (unsigned short*)alloc((size_t)B_ * H_ * 2)};
    float* cC  = (float*)alloc((size_t)B_ * H_ * 4);
    float* chC = (float*)alloc((size_t)B_ * H_ * 4);
    float* cV  = (float*)alloc((size_t)B_ * H_ * 4);
    float* chV = (float*)alloc((size_t)B_ * H_ * 4);
    float* pCb = (float*)alloc((size_t)T_ * B_ * 4);
    float* pVb = (float*)alloc((size_t)T_ * B_ * 4);

    // ---- one-time prep ----
    {
        int n = T_ * B_ * E_;
        cvt_x_kernel<<<(n + 255) / 256, 256, 0, stream>>>(x, Xbf, n);
    }
    auto pack = [&](const float* W, unsigned short* o, int K, int N) {
        pack_b_kernel<<<(K * N + 255) / 256, 256, 0, stream>>>(W, o, K, N);
    };
    pack(Wx_c, WxPkC, E_, G7);
    pack(Wx_v, WxPkV, E_, G7);
    pack(Wh_c, WhPkC, H_, G7);
    pack(Wh_v, WhPkV, H_, G7);
    pack(pcW0, W0PkC, H_, P0_);
    pack(pvW0, W0PkV, H_, P0_);
    pack(pcW1, W1PkC, P0_, P1_);
    pack(pvW1, W1PkV, P0_, P1_);
    zero_state_kernel<<<(B_ * H_ + 255) / 256, 256, 0, stream>>>(
        hC[0], hV[0], cC, chC, cV, chV);

    // ---- recurrence ----
    const size_t head_lds = (size_t)B_ * P0_ * 2 + (size_t)B_ * P1_ * 2; // 96 KB
    for (int t = 0; t < T_; ++t) {
        const int pi = t & 1, po = (t + 1) & 1;
        cell_kernel<<<dim3(H_ / 16, 4, 2), 128, 0, stream>>>(
            Xbf, WxPkC, WhPkC, bx_c, WxPkV, WhPkV, bx_v,
            hC[pi], hC[po], cC, chC,
            hV[pi], hV[po], cV, chV, t);
        head_kernel<<<dim3(2), 256, head_lds, stream>>>(
            hC[po], hV[po],
            W0PkC, pcb0, W1PkC, pcb1, fccW, fccb,
            W0PkV, pvb0, W1PkV, pvb1, fcvW, fcvb,
            pCb, pVb, t);
    }
    finalize_kernel<<<(T_ * B_ + 255) / 256, 256, 0, stream>>>(pCb, pVb, out);
    (void)in_sizes; (void)n_in; (void)out_size; (void)ws_size;
}
